// LinearConv_25323127177743
// MI455X (gfx1250) — matrix operand order repaired and checked by reading, not compiled, not run on hardware
//
#include <hip/hip_runtime.h>
#include <hip/hip_bf16.h>

// ---------------------------------------------------------------------------
// LinearAttention block for MI455X (gfx1250, wave32, bf16 WMMA 16x16x32)
// B=16, C=256, H=W=64 (n=4096), HEADS=4, DIM_HEAD=32, hidden=128, 3*hidden=384
// ---------------------------------------------------------------------------

typedef __bf16 bf16;
typedef __attribute__((ext_vector_type(16))) __bf16 v16bf;
typedef __attribute__((ext_vector_type(8)))  __bf16 v8bf;
typedef __attribute__((ext_vector_type(4)))  __bf16 v4bf;
typedef __attribute__((ext_vector_type(8)))  float  v8f;
typedef __attribute__((ext_vector_type(4)))  float  v4f;

#define B_   16
#define C_   256
#define HWN  4096
#define HID  128
#define OQKV 384
#define NHEAD 4
#define DH   32
#define PER_BATCH (C_ * HWN)     // 1048576
#define SCALE_Q 0.17677669529663687f  // 32^-0.5
#define EPS 1e-5f
#define LDSP 136                 // padded LDS row stride (68 dwords, conflict-free)

// ---------------------------------------------------------------------------
// A-operand fragment (16-bit A 16xK, ISA 7.12.2 table):
//   lane l: row = l&15 ; kb = (l>>4)*8 ; K in [kb,kb+8) U [kb+16,kb+24)
// ---------------------------------------------------------------------------
__device__ __forceinline__ v16bf load_fragA(const bf16* __restrict__ base, int ld) {
  const int lane = threadIdx.x & 31;
  const int r    = lane & 15;
  const int kb   = (lane >> 4) << 3;     // 0 or 8
  const bf16* p  = base + (size_t)r * ld + kb;
  v8bf lo = *(const v8bf*)(p);
  v8bf hi = *(const v8bf*)(p + 16);
  v16bf v;
#pragma unroll
  for (int i = 0; i < 8; ++i) { v[i] = lo[i]; v[i + 8] = hi[i]; }
  return v;
}

// ---------------------------------------------------------------------------
// B-operand fragment (16-bit B Kx16, same K order as A fragment):
//   lane l: col = l&15 ; kb = (l>>4)*8 ; K in [kb,kb+8) U [kb+16,kb+24)
//   (storage must be column-contiguous)
// ---------------------------------------------------------------------------
__device__ __forceinline__ v16bf load_fragB(const bf16* __restrict__ base, int ld) {
  const int lane = threadIdx.x & 31;
  const int col  = lane & 15;
  const int kb   = (lane >> 4) << 3;     // 0 or 8
  const bf16* p  = base + (size_t)col * ld + kb;
  v8bf lo = *(const v8bf*)(p);
  v8bf hi = *(const v8bf*)(p + 16);
  v16bf v;
#pragma unroll
  for (int i = 0; i < 8; ++i) { v[i] = lo[i]; v[i + 8] = hi[i]; }
  return v;
}

__device__ __forceinline__ v8f wmma_bf16(v16bf a, v16bf b, v8f c) {
  return __builtin_amdgcn_wmma_f32_16x16x32_bf16(false, a, false, b,
                                                 (short)0, c, false, false);
}

__device__ __forceinline__ void block_reduce_2(float& s, float& q) {
  __shared__ float rs[256];
  __shared__ float rq[256];
  const int t = threadIdx.x;
  rs[t] = s; rq[t] = q;
  __syncthreads();
#pragma unroll
  for (int off = 128; off > 0; off >>= 1) {
    if (t < off) { rs[t] += rs[t + off]; rq[t] += rq[t + off]; }
    __syncthreads();
  }
  s = rs[0]; q = rq[0];
}

// ---------------------------------------------------------------------------
// K1: per-batch sum / sum-of-squares (float4 loads, LDS + atomic reduction)
// grid (64, B), block 256
// ---------------------------------------------------------------------------
__global__ void gn_stats(const float* __restrict__ X, float* __restrict__ sum,
                         float* __restrict__ ssq) {
  const int b = blockIdx.y;
  const v4f* p = (const v4f*)(X + (size_t)b * PER_BATCH);
  float s = 0.f, q = 0.f;
  for (int i = blockIdx.x * 256 + threadIdx.x; i < PER_BATCH / 4; i += 64 * 256) {
    v4f v = p[i];
#pragma unroll
    for (int j = 0; j < 4; ++j) { s += v[j]; q += v[j] * v[j]; }
  }
  block_reduce_2(s, q);
  if (threadIdx.x == 0) { atomicAdd(&sum[b], s); atomicAdd(&ssq[b], q); }
}

__global__ void gn_finalize(const float* __restrict__ sum, const float* __restrict__ ssq,
                            float* __restrict__ mean, float* __restrict__ rstd) {
  const int b = threadIdx.x;
  if (b < B_) {
    const float inv_n = 1.f / (float)PER_BATCH;
    float mu  = sum[b] * inv_n;
    float var = ssq[b] * inv_n - mu * mu;
    mean[b] = mu;
    rstd[b] = rsqrtf(var + EPS);
  }
}

// ---------------------------------------------------------------------------
// Weights -> bf16 (tiny)
// ---------------------------------------------------------------------------
__global__ void cvt_weights(const float* __restrict__ Win, const float* __restrict__ Wout,
                            bf16* __restrict__ WinB, bf16* __restrict__ WoutB) {
  const int i = blockIdx.x * 256 + threadIdx.x;
  if (i < OQKV * C_)  WinB[i]  = (__bf16)Win[i];
  if (i < C_ * HID)   WoutB[i] = (__bf16)Wout[i];
}

// ---------------------------------------------------------------------------
// K2: GN1 normalize + bf16 + transpose -> Xn_t[b][hw][c]  (c contiguous)
// grid (HWN/64, C_/64, B), block 256
// ---------------------------------------------------------------------------
__global__ void gn1_norm_transpose(const float* __restrict__ X,
                                   const float* __restrict__ w, const float* __restrict__ bb,
                                   const float* __restrict__ mean, const float* __restrict__ rstd,
                                   bf16* __restrict__ Xnt) {
  __shared__ bf16 tile[64][66];     // [c_local][hw_local], odd-dword row stride
  const int b  = blockIdx.z;
  const int c0 = blockIdx.y * 64;
  const int n0 = blockIdx.x * 64;
  const float mu = mean[b], rs = rstd[b];
  const int tx = threadIdx.x & 63;   // hw (read) / c (write)
  const int ty = threadIdx.x >> 6;   // 0..3
#pragma unroll
  for (int i = 0; i < 16; ++i) {
    const int cl = ty + i * 4;
    const int c  = c0 + cl;
    float v = X[((size_t)b * C_ + c) * HWN + n0 + tx];
    v = (v - mu) * rs * w[c] + bb[c];
    tile[cl][tx] = (__bf16)v;
  }
  __syncthreads();
#pragma unroll
  for (int i = 0; i < 16; ++i) {
    const int nl = ty + i * 4;
    Xnt[((size_t)b * HWN + n0 + nl) * C_ + c0 + tx] = tile[tx][nl];
  }
}

// ---------------------------------------------------------------------------
// K3: qkv GEMM  (384 x 4096 per batch, K=256), one wave per 16x64 tile,
//     software-pipelined k-loop (fragments for k+32 issued before k WMMAs).
// Q rows (o<128)  -> Qbuf[b][n][o]  (v8f contiguous stores)
// K/V rows        -> KV[b][o-128][n]
// grid (HWN/64, OQKV/16, B), block 32
// ---------------------------------------------------------------------------
__global__ void gemm_qkv(const bf16* __restrict__ WinB, const bf16* __restrict__ Xnt,
                         float* __restrict__ Qbuf, float* __restrict__ KV) {
  const int nb = blockIdx.x, mt = blockIdx.y, b = blockIdx.z;
  const int lane = threadIdx.x & 31;
  const int m0 = mt * 16, n0 = nb * 64;
  const bf16* Arow  = WinB + (size_t)m0 * C_;
  const bf16* Bbase = Xnt + ((size_t)b * HWN + n0) * C_;
  v8f acc[4] = {};
  v16bf a = load_fragA(Arow, C_);
  v16bf bb[4];
#pragma unroll
  for (int j = 0; j < 4; ++j) bb[j] = load_fragB(Bbase + (size_t)j * 16 * C_, C_);
#pragma unroll
  for (int k = 0; k < C_; k += 32) {
    v16bf an;
    v16bf bn[4];
    if (k + 32 < C_) {
      an = load_fragA(Arow + k + 32, C_);
#pragma unroll
      for (int j = 0; j < 4; ++j)
        bn[j] = load_fragB(Bbase + (size_t)j * 16 * C_ + k + 32, C_);
    }
#pragma unroll
    for (int j = 0; j < 4; ++j) acc[j] = wmma_bf16(a, bb[j], acc[j]);
    if (k + 32 < C_) {
      a = an;
#pragma unroll
      for (int j = 0; j < 4; ++j) bb[j] = bn[j];
    }
  }
  const int cl = lane & 15;
  const int rb = (lane >> 4) << 3;
  if (m0 < HID) {                       // Q region -> [b][n][o]
#pragma unroll
    for (int j = 0; j < 4; ++j) {
      float* dst = Qbuf + ((size_t)b * HWN + n0 + j * 16 + cl) * HID + m0 + rb;
      *(v8f*)dst = acc[j];
    }
  } else {                              // K/V region -> [b][ro][n]
    const int ro = m0 - HID;
#pragma unroll
    for (int j = 0; j < 4; ++j)
#pragma unroll
      for (int r = 0; r < 8; ++r)
        KV[((size_t)b * 256 + ro + rb + r) * HWN + n0 + j * 16 + cl] = acc[j][r];
  }
}

// ---------------------------------------------------------------------------
// K4a: Q softmax over d (32 contiguous floats per (b,n,h)), * SCALE, -> bf16
// grid 1024, block 256 ; total threads = B*HWN*NHEAD = 262144
// ---------------------------------------------------------------------------
__global__ void q_softmax(const float* __restrict__ Qbuf, bf16* __restrict__ Qs) {
  const int idx = blockIdx.x * 256 + threadIdx.x;
  const float* p = Qbuf + (size_t)idx * 32;
  v4f v[8];
  float mx = -3.4e38f;
#pragma unroll
  for (int i = 0; i < 8; ++i) {
    v[i] = *(const v4f*)(p + i * 4);
#pragma unroll
    for (int j = 0; j < 4; ++j) mx = fmaxf(mx, v[i][j]);
  }
  float s = 0.f;
#pragma unroll
  for (int i = 0; i < 8; ++i)
#pragma unroll
    for (int j = 0; j < 4; ++j) { v[i][j] = __expf(v[i][j] - mx); s += v[i][j]; }
  const float sc = SCALE_Q / s;
  bf16* q = Qs + (size_t)idx * 32;
#pragma unroll
  for (int i = 0; i < 4; ++i) {
    v8bf o;
#pragma unroll
    for (int j = 0; j < 4; ++j) {
      o[j]     = (__bf16)(v[2 * i][j] * sc);
      o[4 + j] = (__bf16)(v[2 * i + 1][j] * sc);
    }
    *(v8bf*)(q + i * 8) = o;
  }
}

// ---------------------------------------------------------------------------
// K4b: K softmax over n (rows 0..127) -> Ks ; V convert (rows 128..255) -> Vs
// grid B*256 blocks (one row each), block 256
// ---------------------------------------------------------------------------
__global__ void k_softmax_vcvt(const float* __restrict__ KV, bf16* __restrict__ Ks,
                               bf16* __restrict__ Vs) {
  const int row = blockIdx.x;
  const int b = row >> 8, ro = row & 255;
  const float* p = KV + (size_t)row * HWN;
  const int t = threadIdx.x;
  if (ro >= HID) {                         // V row: just convert
    bf16* dst = Vs + ((size_t)b * HID + (ro - HID)) * HWN;
    for (int i = t * 4; i < HWN; i += 1024) {
      v4f v = *(const v4f*)(p + i);
      v4bf o;
#pragma unroll
      for (int j = 0; j < 4; ++j) o[j] = (__bf16)v[j];
      *(v4bf*)(dst + i) = o;
    }
    return;
  }
  __shared__ float red[256];
  float e[16];
  float mx = -3.4e38f;
#pragma unroll
  for (int i = 0; i < 16; ++i) { e[i] = p[t + i * 256]; mx = fmaxf(mx, e[i]); }
  red[t] = mx; __syncthreads();
#pragma unroll
  for (int off = 128; off > 0; off >>= 1) {
    if (t < off) red[t] = fmaxf(red[t], red[t + off]);
    __syncthreads();
  }
  mx = red[0]; __syncthreads();
  float s = 0.f;
#pragma unroll
  for (int i = 0; i < 16; ++i) { e[i] = __expf(e[i] - mx); s += e[i]; }
  red[t] = s; __syncthreads();
#pragma unroll
  for (int off = 128; off > 0; off >>= 1) {
    if (t < off) red[t] += red[t + off];
    __syncthreads();
  }
  const float inv = 1.f / red[0];
  bf16* dst = Ks + (size_t)row * HWN;       // rows 0..127 == h*32+d
#pragma unroll
  for (int i = 0; i < 16; ++i) dst[t + i * 256] = (__bf16)(e[i] * inv);
}

// ---------------------------------------------------------------------------
// K5: context[d,e] = sum_n Ks[d,n]*Vs[e,n]  -- 32x32 per (b,h), K-dim 4096,
//     software-pipelined. Output stored pre-transposed ctxT[b][h][e][d] bf16.
// grid 64, block 32
// ---------------------------------------------------------------------------
__global__ void ctx_gemm(const bf16* __restrict__ Ks, const bf16* __restrict__ Vs,
                         bf16* __restrict__ CtxT) {
  const int bh = blockIdx.x;
  const int lane = threadIdx.x & 31;
  const bf16* Kh = Ks + (size_t)bh * DH * HWN;
  const bf16* Vh = Vs + (size_t)bh * DH * HWN;
  v8f acc[2][2] = {};
  v16bf a0 = load_fragA(Kh, HWN);
  v16bf a1 = load_fragA(Kh + (size_t)16 * HWN, HWN);
  v16bf b0 = load_fragB(Vh, HWN);
  v16bf b1 = load_fragB(Vh + (size_t)16 * HWN, HWN);
  for (int k = 32; k < HWN; k += 32) {
    v16bf na0 = load_fragA(Kh + k, HWN);
    v16bf na1 = load_fragA(Kh + (size_t)16 * HWN + k, HWN);
    v16bf nb0 = load_fragB(Vh + k, HWN);
    v16bf nb1 = load_fragB(Vh + (size_t)16 * HWN + k, HWN);
    acc[0][0] = wmma_bf16(a0, b0, acc[0][0]);
    acc[0][1] = wmma_bf16(a0, b1, acc[0][1]);
    acc[1][0] = wmma_bf16(a1, b0, acc[1][0]);
    acc[1][1] = wmma_bf16(a1, b1, acc[1][1]);
    a0 = na0; a1 = na1; b0 = nb0; b1 = nb1;
  }
  acc[0][0] = wmma_bf16(a0, b0, acc[0][0]);
  acc[0][1] = wmma_bf16(a0, b1, acc[0][1]);
  acc[1][0] = wmma_bf16(a1, b0, acc[1][0]);
  acc[1][1] = wmma_bf16(a1, b1, acc[1][1]);
  const int cl = lane & 15;
  const int rb = (lane >> 4) << 3;
#pragma unroll
  for (int mi = 0; mi < 2; ++mi)
#pragma unroll
    for (int jn = 0; jn < 2; ++jn) {
      v8bf o;
#pragma unroll
      for (int r = 0; r < 8; ++r) o[r] = (__bf16)acc[mi][jn][r];
      bf16* dst = CtxT + ((size_t)bh * DH + jn * 16 + cl) * DH + mi * 16 + rb;
      *(v8bf*)dst = o;                 // ctxT[e][d], d contiguous
    }
}

// ---------------------------------------------------------------------------
// K6: out[e,n] = ctxT @ Qs (K=32) staged in LDS, then y = W_out @ out (K=128)
//     + b_out.  Writes ypre[b][o][n] and accumulates GN2 sum/sumsq.
// grid (HWN/64, 1, B), block 256 (8 waves)
// ---------------------------------------------------------------------------
__global__ void out_proj(const bf16* __restrict__ CtxT, const bf16* __restrict__ Qs,
                         const bf16* __restrict__ WoutB, const float* __restrict__ b_out,
                         float* __restrict__ ypre, float* __restrict__ sum2,
                         float* __restrict__ ssq2) {
  __shared__ bf16 outb[64 * LDSP];     // [n_local][k=h*32+e], padded rows
  const int b = blockIdx.z;
  const int n0 = blockIdx.x * 64;
  const int wave = threadIdx.x >> 5;
  const int lane = threadIdx.x & 31;
  const int cl = lane & 15;
  const int rb = (lane >> 4) << 3;

  // ---- stage A: 8x4 = 32 tiles of out (16 he x 16 n), one WMMA each ----
#pragma unroll
  for (int tI = wave * 4; tI < wave * 4 + 4; ++tI) {
    const int mt = tI >> 2;            // he-tile 0..7
    const int nt = tI & 3;             // n-tile 0..3
    const int h  = mt >> 1;
    const int el = (mt & 1) * 16;
    v16bf a  = load_fragA(CtxT + ((size_t)(b * NHEAD + h) * DH + el) * DH, DH);
    v16bf bf = load_fragB(Qs + ((size_t)b * HWN + n0 + nt * 16) * HID + h * DH, HID);
    v8f z = {};
    v8f d = wmma_bf16(a, bf, z);
    v8bf o;
#pragma unroll
    for (int r = 0; r < 8; ++r) o[r] = (__bf16)d[r];
    *(v8bf*)&outb[(nt * 16 + cl) * LDSP + mt * 16 + rb] = o;
  }
  __syncthreads();

  // ---- stage B: 16x4 = 64 tiles of y (16 o x 16 n), K=128 ----
  float ls = 0.f, lq = 0.f;
#pragma unroll
  for (int tI = wave * 8; tI < wave * 8 + 8; ++tI) {
    const int mt = tI >> 2;            // o-tile 0..15
    const int nt = tI & 3;
    v8f acc = {};
#pragma unroll
    for (int k = 0; k < HID; k += 32) {
      v16bf a  = load_fragA(WoutB + (size_t)mt * 16 * HID + k, HID);
      v16bf bf = load_fragB(&outb[nt * 16 * LDSP + k], LDSP);
      acc = wmma_bf16(a, bf, acc);
    }
    const int o0 = mt * 16 + rb;
#pragma unroll
    for (int r = 0; r < 8; ++r) {
      float v = acc[r] + b_out[o0 + r];
      ypre[((size_t)b * C_ + o0 + r) * HWN + n0 + nt * 16 + cl] = v;
      ls += v; lq += v * v;
    }
  }
  block_reduce_2(ls, lq);
  if (threadIdx.x == 0) { atomicAdd(&sum2[b], ls); atomicAdd(&ssq2[b], lq); }
}

// ---------------------------------------------------------------------------
// K7: out = (ypre - mu2)*rstd2*gn2_w + gn2_b + X   (float4 vectorized)
// grid 16384, block 256
// ---------------------------------------------------------------------------
__global__ void gn2_residual(const float* __restrict__ ypre, const float* __restrict__ X,
                             const float* __restrict__ w, const float* __restrict__ bb,
                             const float* __restrict__ mean2, const float* __restrict__ rstd2,
                             float* __restrict__ out) {
  const size_t i = ((size_t)blockIdx.x * 256 + threadIdx.x) * 4;
  const int b = (int)(i >> 20);            // / (C_*HWN)
  const int c = (int)((i >> 12) & 255);    // / HWN % C_
  const float mu = mean2[b], rs = rstd2[b];
  const float wc = w[c], bc = bb[c];
  v4f y = *(const v4f*)(ypre + i);
  v4f x = *(const v4f*)(X + i);
  v4f o;
#pragma unroll
  for (int j = 0; j < 4; ++j) o[j] = (y[j] - mu) * rs * wc + bc + x[j];
  *(v4f*)(out + i) = o;
}

// ---------------------------------------------------------------------------
// Host side
// ---------------------------------------------------------------------------
extern "C" void kernel_launch(void* const* d_in, const int* in_sizes, int n_in,
                              void* d_out, int out_size, void* d_ws, size_t ws_size,
                              hipStream_t stream) {
  const float* X    = (const float*)d_in[0];
  const float* gn1w = (const float*)d_in[1];
  const float* gn1b = (const float*)d_in[2];
  const float* Win  = (const float*)d_in[3];
  const float* Wout = (const float*)d_in[4];
  const float* bout = (const float*)d_in[5];
  const float* gn2w = (const float*)d_in[6];
  const float* gn2b = (const float*)d_in[7];
  float* Y = (float*)d_out;

  char* ws = (char*)d_ws;
  // stats block (zeroed each launch)
  float* s1 = (float*)ws;            // [16]
  float* q1 = s1 + 16;
  float* m1 = s1 + 32;
  float* r1 = s1 + 48;
  float* s2 = s1 + 64;
  float* q2 = s1 + 80;
  float* m2 = s1 + 96;
  float* r2 = s1 + 112;
  // big scratch (~185 MB total; ypre aliases dead Xnt+Qbuf region)
  const size_t OFF_WINB  = 1024;
  const size_t OFF_WOUTB = OFF_WINB  + (size_t)OQKV * C_ * 2;        // 197632
  const size_t OFF_CTXT  = OFF_WOUTB + (size_t)C_ * HID * 2;         // 263168
  const size_t OFF_XNT   = OFF_CTXT  + (size_t)B_ * NHEAD * DH * DH * 2; // 394240
  const size_t OFF_QBUF  = OFF_XNT   + (size_t)B_ * HWN * C_ * 2;    // +33.5MB
  const size_t OFF_KV    = OFF_QBUF  + (size_t)B_ * HWN * HID * 4;   // +33.5MB
  const size_t OFF_QS    = OFF_KV    + (size_t)B_ * 256 * HWN * 4;   // +67MB
  const size_t OFF_KS    = OFF_QS    + (size_t)B_ * HWN * HID * 2;
  const size_t OFF_VS    = OFF_KS    + (size_t)B_ * HID * HWN * 2;
  bf16*  WinB  = (bf16*)(ws + OFF_WINB);
  bf16*  WoutB = (bf16*)(ws + OFF_WOUTB);
  bf16*  CtxT  = (bf16*)(ws + OFF_CTXT);
  bf16*  Xnt   = (bf16*)(ws + OFF_XNT);
  float* Qbuf  = (float*)(ws + OFF_QBUF);
  float* KV    = (float*)(ws + OFF_KV);
  bf16*  Qs    = (bf16*)(ws + OFF_QS);
  bf16*  Ks    = (bf16*)(ws + OFF_KS);
  bf16*  Vs    = (bf16*)(ws + OFF_VS);
  float* ypre  = (float*)(ws + OFF_XNT);   // alias: Xnt+Qbuf dead by then

  hipMemsetAsync(d_ws, 0, 512, stream);    // zero stats accumulators

  gn_stats<<<dim3(64, B_), 256, 0, stream>>>(X, s1, q1);
  gn_finalize<<<1, 32, 0, stream>>>(s1, q1, m1, r1);
  cvt_weights<<<(OQKV * C_ + 255) / 256, 256, 0, stream>>>(Win, Wout, WinB, WoutB);
  gn1_norm_transpose<<<dim3(HWN / 64, C_ / 64, B_), 256, 0, stream>>>(
      X, gn1w, gn1b, m1, r1, Xnt);
  gemm_qkv<<<dim3(HWN / 64, OQKV / 16, B_), 32, 0, stream>>>(WinB, Xnt, Qbuf, KV);
  q_softmax<<<(B_ * HWN * NHEAD) / 256, 256, 0, stream>>>(Qbuf, Qs);
  k_softmax_vcvt<<<B_ * 256, 256, 0, stream>>>(KV, Ks, Vs);
  ctx_gemm<<<B_ * NHEAD, 32, 0, stream>>>(Ks, Vs, CtxT);
  out_proj<<<dim3(HWN / 64, 1, B_), 256, 0, stream>>>(CtxT, Qs, WoutB, bout,
                                                      ypre, s2, q2);
  gn_finalize<<<1, 32, 0, stream>>>(s2, q2, m2, r2);
  gn2_residual<<<(B_ * PER_BATCH) / 1024, 256, 0, stream>>>(ypre, X, gn2w, gn2b,
                                                            m2, r2, Y);
}
